// Seq2seq_53944789238574
// MI455X (gfx1250) — compile-verified, matrix operand order checked by reading, not run on hardware
//
#include <hip/hip_runtime.h>
#include <cstdint>
#include <cstddef>

// Problem constants (match reference)
#define B_ 32
#define S_ 64
#define T_ 64
#define E_ 512
#define H_ 512
#define L_ 4
#define V_ 32003

typedef __attribute__((ext_vector_type(16))) __bf16 v16bf;
typedef __attribute__((ext_vector_type(8)))  float  v8f;

union FragU { v16bf v; uint4 u[2]; };

// A/B fragment loader for V_WMMA_F32_16X16X32_BF16.
// Per ISA layout: lanes 0-15 hold K chunks {0..7, 16..23}, lanes 16-31 hold
// {8..15, 24..31} for the same 16 rows/cols. Caller passes the per-lane base
// pointer already offset to (row, k + half*8); we grab 8 bf16 + 8 bf16 (+16).
__device__ __forceinline__ v16bf load_frag(const unsigned short* p) {
  FragU f;
  f.u[0] = *reinterpret_cast<const uint4*>(p);
  f.u[1] = *reinterpret_cast<const uint4*>(p + 16);
  return f.v;
}

__device__ __forceinline__ v8f wmma_bf16(v16bf a, v16bf b, v8f c) {
  return __builtin_amdgcn_wmma_f32_16x16x32_bf16(
      /*neg_a=*/false, a, /*neg_b=*/false, b,
      /*c_mod=*/(short)0, c, /*reuse_a=*/false, /*reuse_b=*/false);
}

__device__ __forceinline__ unsigned short f2bf(float x) {
  unsigned int u = __float_as_uint(x);
  u = (u + 0x7FFFu + ((u >> 16) & 1u)) >> 16;  // round-to-nearest-even
  return (unsigned short)u;
}

__device__ __forceinline__ float sigmoidf_(float x) {
  return 1.0f / (1.0f + __expf(-x));
}

// ---------------------------------------------------------------- utilities
__global__ void cvt_f32_bf16_kernel(const float* __restrict__ src,
                                    unsigned short* __restrict__ dst, int n) {
  int i = blockIdx.x * blockDim.x + threadIdx.x;
  int stride = gridDim.x * blockDim.x;
  for (; i < n; i += stride) dst[i] = f2bf(src[i]);
}

__global__ void zero_f32_kernel(float* p, int n) {
  int i = blockIdx.x * blockDim.x + threadIdx.x;
  int stride = gridDim.x * blockDim.x;
  for (; i < n; i += stride) p[i] = 0.0f;
}

__global__ void zero_u16_kernel(unsigned short* p, int n) {
  int i = blockIdx.x * blockDim.x + threadIdx.x;
  int stride = gridDim.x * blockDim.x;
  for (; i < n; i += stride) p[i] = 0;
}

// ------------------------------------------------------- generic bf16 GEMM
// out[m,n] = bf16( sum_k A[m,k]*W[n,k] + bias[n] )   (A:[M,K] lda, W:[N,K] ldw)
// grid = (N/16, M/16), block = 32 (one wave per 16x16 tile)
__global__ void gemm_bf16_kernel(const unsigned short* __restrict__ A, int lda,
                                 const unsigned short* __restrict__ W, int ldw,
                                 const float* __restrict__ bias,
                                 unsigned short* __restrict__ out, int ldo, int K) {
  int tn = blockIdx.x, tm = blockIdx.y;
  int lane = threadIdx.x & 31;
  int half = lane >> 4, lm = lane & 15;
  const unsigned short* arow = A + (size_t)(tm * 16 + lm) * lda;
  const unsigned short* wrow = W + (size_t)(tn * 16 + lm) * ldw;
  v8f acc = {};
  for (int k = 0; k < K; k += 32) {
    int k0 = k + half * 8;
    v16bf a = load_frag(arow + k0);
    v16bf b = load_frag(wrow + k0);
    acc = wmma_bf16(a, b, acc);
  }
  int col = tn * 16 + lm;
  float bv = bias ? bias[col] : 0.0f;
  for (int v = 0; v < 8; ++v) {
    int row = tm * 16 + v + 8 * half;  // C layout: VGPR v, half selects M+8
    out[(size_t)row * ldo + col] = f2bf(acc[v] + bv);
  }
}

// ------------------------------------------------- fused GRU layer-step
// One kernel = one (time-step, layer). grid = H/16 column tiles, block = 64
// (wave 0 -> batch rows 0..15, wave 1 -> rows 16..31). Each wave accumulates
// the 6 gate tiles (gi_r/z/n = x@Wih^T, gh_r/z/n = h@Whh^T) with WMMA, then
// applies the GRU gate math in fp32 and writes h (f32 state + bf16 copy).
// Double-buffered h (read pr / write pw) avoids cross-WG hazards on gh inputs.
__global__ void gru_layer_kernel(const unsigned short* __restrict__ x, int ldx,
                                 const unsigned short* __restrict__ hprev_bf,
                                 const float* __restrict__ hprev_f,
                                 const unsigned short* __restrict__ Wih,
                                 const unsigned short* __restrict__ Whh,
                                 const float* __restrict__ bih,
                                 const float* __restrict__ bhh,
                                 unsigned short* __restrict__ hnew_bf,
                                 float* __restrict__ hnew_f,
                                 float* __restrict__ eo_out, int ldeo) {
  int tn = blockIdx.x;                 // column tile within H
  int wv = threadIdx.x >> 5;           // M-tile (0 or 1)
  int lane = threadIdx.x & 31;
  int half = lane >> 4, lm = lane & 15;

  const unsigned short* xrow = x + (size_t)(wv * 16 + lm) * ldx;
  const unsigned short* hrow = hprev_bf + (size_t)(wv * 16 + lm) * H_;
  int nloc = tn * 16 + lm;             // output column handled by this lane (B frag)
  const unsigned short* wr = Wih + (size_t)(0 * H_ + nloc) * H_;
  const unsigned short* wz = Wih + (size_t)(1 * H_ + nloc) * H_;
  const unsigned short* wn = Wih + (size_t)(2 * H_ + nloc) * H_;
  const unsigned short* ur = Whh + (size_t)(0 * H_ + nloc) * H_;
  const unsigned short* uz = Whh + (size_t)(1 * H_ + nloc) * H_;
  const unsigned short* un = Whh + (size_t)(2 * H_ + nloc) * H_;

  v8f gir = {}, giz = {}, gin = {}, ghr = {}, ghz = {}, ghn = {};
  for (int k = 0; k < H_; k += 32) {
    int k0 = k + half * 8;
    v16bf ax = load_frag(xrow + k0);
    gir = wmma_bf16(ax, load_frag(wr + k0), gir);
    giz = wmma_bf16(ax, load_frag(wz + k0), giz);
    gin = wmma_bf16(ax, load_frag(wn + k0), gin);
    v16bf ah = load_frag(hrow + k0);
    ghr = wmma_bf16(ah, load_frag(ur + k0), ghr);
    ghz = wmma_bf16(ah, load_frag(uz + k0), ghz);
    ghn = wmma_bf16(ah, load_frag(un + k0), ghn);
  }

  int col = tn * 16 + lm;
  float bir = bih[col], biz = bih[H_ + col], bin = bih[2 * H_ + col];
  float bhr = bhh[col], bhz = bhh[H_ + col], bhn = bhh[2 * H_ + col];
  for (int v = 0; v < 8; ++v) {
    int row = wv * 16 + v + 8 * half;  // batch index
    float r = sigmoidf_((gir[v] + bir) + (ghr[v] + bhr));
    float z = sigmoidf_((giz[v] + biz) + (ghz[v] + bhz));
    float n = tanhf((gin[v] + bin) + r * (ghn[v] + bhn));
    float hp = hprev_f[(size_t)row * H_ + col];
    float h = (1.0f - z) * n + z * hp;
    hnew_f[(size_t)row * H_ + col] = h;
    hnew_bf[(size_t)row * H_ + col] = f2bf(h);
    if (eo_out) eo_out[(size_t)row * ldeo + col] = h;  // encoder layer-3 output
  }
}

// ------------------------------------------------- decoder attention (VALU)
// w[s] = eo[b,s,:]·out[b,:]; ctx = sum_s w[s]*eo[b,s,:]; cat=[ctx,out] -> bf16
__global__ void attn_kernel(const float* __restrict__ h3,
                            const float* __restrict__ eo,
                            unsigned short* __restrict__ cat_out) {
  __shared__ float outv[H_];
  __shared__ float wsh[S_];
  int b = blockIdx.x;
  int tid = threadIdx.x;  // 256
  for (int c = tid; c < H_; c += 256) outv[c] = h3[(size_t)b * H_ + c];
  __syncthreads();
  int s = tid >> 2, q = tid & 3;  // 4 threads per source step
  const float* er = eo + ((size_t)b * S_ + s) * H_;
  float p = 0.0f;
  for (int c = q; c < H_; c += 4) p += er[c] * outv[c];
  p += __shfl_xor(p, 1, 32);
  p += __shfl_xor(p, 2, 32);
  if (q == 0) wsh[s] = p;
  __syncthreads();
  for (int c = tid; c < H_; c += 256) {
    float acc = 0.0f;
    for (int ss = 0; ss < S_; ++ss)
      acc += wsh[ss] * eo[((size_t)b * S_ + ss) * H_ + c];
    cat_out[(size_t)b * 2 * H_ + c] = f2bf(acc);          // ctx
    cat_out[(size_t)b * 2 * H_ + H_ + c] = f2bf(outv[c]); // out
  }
}

// ---------------------------------- fused vocab GEMM + log-softmax CE + argmax
// grid = (T*B)/16 row tiles, block = 256 (8 waves; wave w owns cols w*16 of a
// 128-wide chunk). cats rows cached in LDS; W_out streamed (bf16, L2-resident).
// Online softmax per row; logits never written to memory.
__global__ void out_kernel(const unsigned short* __restrict__ cats,
                           const unsigned short* __restrict__ Wout,
                           const float* __restrict__ bout,
                           const int* __restrict__ tseq,
                           const float* __restrict__ lmask,
                           float* __restrict__ dout) {
  __shared__ __align__(16) unsigned short aTile[16 * 1024];
  __shared__ float ltile[16 * 128];
  __shared__ float s_max[16], s_sum[16], s_amv[16], s_tgt[16];
  __shared__ int s_ami[16];

  int blk = blockIdx.x;
  int tid = threadIdx.x;

  // Stage the 16 cats rows (16 x 1024 bf16 = 32KB) into LDS.
  for (int f = tid; f < 2048; f += 256) {  // 2048 uint4 = 32KB
    int idx = f * 8;
    int r = idx >> 10, c = idx & 1023;
    *reinterpret_cast<uint4*>(&aTile[idx]) =
        *reinterpret_cast<const uint4*>(cats + ((size_t)(blk * 16 + r)) * 1024 + c);
  }
  if (tid < 16) {
    s_max[tid] = -INFINITY; s_sum[tid] = 0.0f;
    s_amv[tid] = -INFINITY; s_ami[tid] = 0; s_tgt[tid] = 0.0f;
  }
  __syncthreads();

  int wv = tid >> 5, lane = tid & 31, half = lane >> 4, lm = lane & 15;
  const int nchunks = (V_ + 127) / 128;

  for (int cc = 0; cc < nchunks; ++cc) {
    int nrow = cc * 128 + wv * 16 + lm;
    int nclamp = (nrow < V_) ? nrow : 0;
    const unsigned short* wrow = Wout + (size_t)nclamp * 1024;
    const unsigned short* arow = &aTile[(size_t)lm * 1024];
    v8f acc = {};
    for (int k = 0; k < 1024; k += 32) {
      int k0 = k + half * 8;
      v16bf a = load_frag(arow + k0);   // ds_load_b128 x2
      v16bf b = load_frag(wrow + k0);   // global (L2 resident)
      acc = wmma_bf16(a, b, acc);
    }
    for (int v = 0; v < 8; ++v)
      ltile[(v + 8 * half) * 128 + wv * 16 + lm] = acc[v];
    __syncthreads();

    // Online-softmax merge: 16 threads per row (j = lane-in-row).
    int r = tid >> 4, j = tid & 15;
    int rowg = blk * 16 + r;            // rowg = t*B + b
    int t_ = rowg >> 5, b_ = rowg & 31;
    int tgt = tseq[b_ * T_ + t_];
    float vals[8];
    float lmax = -INFINITY; int lidx = -1;
    for (int i = 0; i < 8; ++i) {
      int c = j + i * 16;
      int col = cc * 128 + c;
      float val = (col < V_) ? (ltile[r * 128 + c] + bout[col]) : -INFINITY;
      vals[i] = val;
      if (val > lmax) { lmax = val; lidx = col; }
      if (col == tgt) s_tgt[r] = val;
    }
    for (int off = 8; off >= 1; off >>= 1) {
      float om = __shfl_xor(lmax, off, 16);
      int   oi = __shfl_xor(lidx, off, 16);
      if (om > lmax || (om == lmax && oi < lidx)) { lmax = om; lidx = oi; }
    }
    float lsum = 0.0f;
    for (int i = 0; i < 8; ++i) lsum += __expf(vals[i] - lmax);
    for (int off = 8; off >= 1; off >>= 1) lsum += __shfl_xor(lsum, off, 16);
    if (j == 0) {
      float rm = s_max[r], rs = s_sum[r];
      float nm = fmaxf(rm, lmax);
      s_sum[r] = rs * __expf(rm - nm) + lsum * __expf(lmax - nm);
      s_max[r] = nm;
      if (lmax > s_amv[r]) { s_amv[r] = lmax; s_ami[r] = lidx; }
    }
    __syncthreads();
  }

  if (tid < 16) {
    int rowg = blk * 16 + tid;
    int t_ = rowg >> 5, b_ = rowg & 31;
    float ce = -(s_tgt[tid] - s_max[tid] - __logf(s_sum[tid]));
    atomicAdd(dout, ce * lmask[b_ * T_ + t_] * (1.0f / (float)B_));
    dout[1 + b_ * T_ + t_] = (float)s_ami[tid];  // argmax as float
  }
}

// ---------------------------------------------------------------- launcher
extern "C" void kernel_launch(void* const* d_in, const int* in_sizes, int n_in,
                              void* d_out, int out_size, void* d_ws, size_t ws_size,
                              hipStream_t stream) {
  (void)in_sizes; (void)n_in; (void)out_size; (void)ws_size;
  const float* query   = (const float*)d_in[0];
  const float* targetI = (const float*)d_in[1];
  const int*   tseq    = (const int*)d_in[2];
  const float* lmask   = (const float*)d_in[3];
  const float* encWih  = (const float*)d_in[4];
  const float* encWhh  = (const float*)d_in[5];
  const float* enc_bih = (const float*)d_in[6];
  const float* enc_bhh = (const float*)d_in[7];
  const float* decWih  = (const float*)d_in[8];
  const float* decWhh  = (const float*)d_in[9];
  const float* dec_bih = (const float*)d_in[10];
  const float* dec_bhh = (const float*)d_in[11];
  const float* Wemb    = (const float*)d_in[12];
  const float* b_emb   = (const float*)d_in[13];
  const float* Wout    = (const float*)d_in[14];
  const float* b_out   = (const float*)d_in[15];
  float* out = (float*)d_out;

  char* wsb = (char*)d_ws;
  size_t off = 0;
  auto alloc = [&](size_t bytes) -> void* {
    void* p = wsb + off;
    off = (off + bytes + 255) & ~(size_t)255;
    return p;
  };

  const int Tt = T_ + 1;  // target has T+1 steps
  unsigned short* qin_bf  = (unsigned short*)alloc((size_t)B_ * S_ * E_ * 2);
  unsigned short* tin_bf  = (unsigned short*)alloc((size_t)B_ * Tt * E_ * 2);
  unsigned short* Wemb_bf = (unsigned short*)alloc((size_t)H_ * E_ * 2);
  unsigned short* eWih_bf = (unsigned short*)alloc((size_t)L_ * 3 * H_ * H_ * 2);
  unsigned short* eWhh_bf = (unsigned short*)alloc((size_t)L_ * 3 * H_ * H_ * 2);
  unsigned short* dWih_bf = (unsigned short*)alloc((size_t)L_ * 3 * H_ * H_ * 2);
  unsigned short* dWhh_bf = (unsigned short*)alloc((size_t)L_ * 3 * H_ * H_ * 2);
  unsigned short* Wout_bf = (unsigned short*)alloc((size_t)V_ * 2 * H_ * 2);
  unsigned short* qh_bf   = (unsigned short*)alloc((size_t)B_ * S_ * H_ * 2);
  unsigned short* th_bf   = (unsigned short*)alloc((size_t)B_ * Tt * H_ * 2);
  float* hF[2]; unsigned short* hB[2];
  hF[0] = (float*)alloc((size_t)L_ * B_ * H_ * 4);
  hF[1] = (float*)alloc((size_t)L_ * B_ * H_ * 4);
  hB[0] = (unsigned short*)alloc((size_t)L_ * B_ * H_ * 2);
  hB[1] = (unsigned short*)alloc((size_t)L_ * B_ * H_ * 2);
  float* eo = (float*)alloc((size_t)B_ * S_ * H_ * 4);
  unsigned short* cats_bf = (unsigned short*)alloc((size_t)T_ * B_ * 2 * H_ * 2);

  // ---- 1) convert inputs/weights to bf16 ----
  auto cvt = [&](const float* s, unsigned short* d, int n) {
    cvt_f32_bf16_kernel<<<1024, 256, 0, stream>>>(s, d, n);
  };
  cvt(query,   qin_bf,  B_ * S_ * E_);
  cvt(targetI, tin_bf,  B_ * Tt * E_);
  cvt(Wemb,    Wemb_bf, H_ * E_);
  cvt(encWih,  eWih_bf, L_ * 3 * H_ * H_);
  cvt(encWhh,  eWhh_bf, L_ * 3 * H_ * H_);
  cvt(decWih,  dWih_bf, L_ * 3 * H_ * H_);
  cvt(decWhh,  dWhh_bf, L_ * 3 * H_ * H_);
  cvt(Wout,    Wout_bf, V_ * 2 * H_);

  // ---- 2) init state / loss ----
  zero_f32_kernel<<<64, 256, 0, stream>>>(hF[0], L_ * B_ * H_);
  zero_u16_kernel<<<64, 256, 0, stream>>>(hB[0], L_ * B_ * H_);
  zero_f32_kernel<<<1, 32, 0, stream>>>(out, 1);

  // ---- 3) embed projections (WMMA GEMMs) ----
  gemm_bf16_kernel<<<dim3(H_ / 16, (B_ * S_) / 16), 32, 0, stream>>>(
      qin_bf, E_, Wemb_bf, E_, b_emb, qh_bf, H_, E_);
  gemm_bf16_kernel<<<dim3(H_ / 16, (B_ * Tt) / 16), 32, 0, stream>>>(
      tin_bf, E_, Wemb_bf, E_, b_emb, th_bf, H_, E_);

  // ---- 4) encoder scan ----
  int g = 0;
  for (int s = 0; s < S_; ++s, ++g) {
    int pr = g & 1, pw = (g + 1) & 1;
    for (int l = 0; l < L_; ++l) {
      const unsigned short* x; int ldx;
      if (l == 0) { x = qh_bf + (size_t)s * H_; ldx = S_ * H_; }
      else        { x = hB[pw] + (size_t)(l - 1) * B_ * H_; ldx = H_; }
      float* eo_p = (l == L_ - 1) ? (eo + (size_t)s * H_) : nullptr;
      gru_layer_kernel<<<H_ / 16, 64, 0, stream>>>(
          x, ldx,
          hB[pr] + (size_t)l * B_ * H_, hF[pr] + (size_t)l * B_ * H_,
          eWih_bf + (size_t)l * 3 * H_ * H_, eWhh_bf + (size_t)l * 3 * H_ * H_,
          enc_bih + l * 3 * H_, enc_bhh + l * 3 * H_,
          hB[pw] + (size_t)l * B_ * H_, hF[pw] + (size_t)l * B_ * H_,
          eo_p, S_ * H_);
    }
  }

  // ---- 5) decoder scan (continues h parity chain from encoder) ----
  for (int t = 0; t < T_; ++t, ++g) {
    int pr = g & 1, pw = (g + 1) & 1;
    for (int l = 0; l < L_; ++l) {
      const unsigned short* x; int ldx;
      if (l == 0) { x = th_bf + (size_t)t * H_; ldx = Tt * H_; }
      else        { x = hB[pw] + (size_t)(l - 1) * B_ * H_; ldx = H_; }
      gru_layer_kernel<<<H_ / 16, 64, 0, stream>>>(
          x, ldx,
          hB[pr] + (size_t)l * B_ * H_, hF[pr] + (size_t)l * B_ * H_,
          dWih_bf + (size_t)l * 3 * H_ * H_, dWhh_bf + (size_t)l * 3 * H_ * H_,
          dec_bih + l * 3 * H_, dec_bhh + l * 3 * H_,
          hB[pw] + (size_t)l * B_ * H_, hF[pw] + (size_t)l * B_ * H_,
          nullptr, 0);
    }
    attn_kernel<<<B_, 256, 0, stream>>>(
        hF[(g + 1) & 1] + (size_t)(L_ - 1) * B_ * H_, eo,
        cats_bf + (size_t)t * B_ * 2 * H_);
  }

  // ---- 6) fused vocab projection + CE + argmax ----
  out_kernel<<<(T_ * B_) / 16, 256, 0, stream>>>(
      cats_bf, Wout_bf, b_out, tseq, lmask, out);
}
